// NS_ec_3221225472203
// MI455X (gfx1250) — matrix operand order repaired and checked by reading, not compiled, not run on hardware
//
#include <hip/hip_runtime.h>
#include <hip/hip_bf16.h>

// ---------------------------------------------------------------------------
// MI455X (gfx1250) implementation.
//  - Dense matmuls use v_wmma_f32_16x16x32_bf16 (bf16 A/B, fp32 accumulate).
//  - Projection + MLP + softmax are FUSED per 16-row strip: the projection's
//    16x256 output strip is exactly the K-extent the MLP layer-1 needs, so it
//    is staged in LDS as bf16 (never round-tripped through HBM).
//  - Weights are pre-transposed to bf16 (N x K) once per call so each lane's
//    B fragment (K = hi*8+0..7 and 16+hi*8+0..7, as for A) is two b128 loads.
//  - Fragment layouts per CDNA5 ISA:
//      A 16x32 bf16 : lane (hi=l>>4, m=l&15); elems 0..7 -> K=hi*8+e,
//                     elems 8..15 -> K=16+hi*8+(e-8)
//      B 32x16 bf16 : lane (hi, n=l&15); elems 0..7 -> K=hi*8+e, 8..15 -> K=16+hi*8+(e-8)
//      C/D 16x16 f32: lane n=l&15; vgpr r -> row = r + 8*hi
//  - Edge softmax/scatter is atomic-bound; only dst < NODE_NUM edges
//    contribute to the output, so the rest are skipped.
// ---------------------------------------------------------------------------

typedef __attribute__((ext_vector_type(16))) __bf16 v16bf;
typedef __attribute__((ext_vector_type(8)))  __bf16 v8bf;
typedef __attribute__((ext_vector_type(8)))  float  v8f;

#define HIDDEN      256
#define NUM_CLASSES 16
#define N_NODES     100000
#define NODE_NUM    40000
#define N_EDGES     3200000

#define HPAD 264   // bf16 elems per LDS row of h-strip (512B + 16B pad)
#define TPAD 260   // f32  elems per LDS row of t-strip (1024B + 16B pad)

// ---- WMMA fragment loaders ------------------------------------------------

// A fragment from fp32 row-major activations (convert to bf16 in-register).
static __device__ __forceinline__ v16bf load_a_frag_f32(const float* __restrict__ row,
                                                        int k0, int hi) {
  const float4* p0 = (const float4*)(row + k0 + hi * 8);        // K = hi*8 + 0..7
  const float4* p1 = (const float4*)(row + k0 + 16 + hi * 8);   // K = 16+hi*8+0..7
  float4 a0 = p0[0], a1 = p0[1], a2 = p1[0], a3 = p1[1];
  v16bf a;
  a[0]  = (__bf16)a0.x; a[1]  = (__bf16)a0.y; a[2]  = (__bf16)a0.z; a[3]  = (__bf16)a0.w;
  a[4]  = (__bf16)a1.x; a[5]  = (__bf16)a1.y; a[6]  = (__bf16)a1.z; a[7]  = (__bf16)a1.w;
  a[8]  = (__bf16)a2.x; a[9]  = (__bf16)a2.y; a[10] = (__bf16)a2.z; a[11] = (__bf16)a2.w;
  a[12] = (__bf16)a3.x; a[13] = (__bf16)a3.y; a[14] = (__bf16)a3.z; a[15] = (__bf16)a3.w;
  return a;
}

// A fragment from a bf16 LDS row (two ds_load_b128).
static __device__ __forceinline__ v16bf load_a_frag_lds(const __bf16* row, int k0, int hi) {
  const v8bf* p0 = (const v8bf*)(row + k0 + hi * 8);
  const v8bf* p1 = (const v8bf*)(row + k0 + 16 + hi * 8);
  v8bf a0 = p0[0];
  v8bf a1 = p1[0];
  v16bf a;
#pragma unroll
  for (int e = 0; e < 8; ++e) { a[e] = a0[e]; a[8 + e] = a1[e]; }
  return a;
}

// B fragment from pre-transposed bf16 weights Wt (N x K row-major):
// lane column n needs K = k0 + hi*8 .. +7 and k0 + 16 + hi*8 .. +7 (two b128).
static __device__ __forceinline__ v16bf load_b_frag_bf16(const __bf16* __restrict__ Wt,
                                                         int K, int k0, int n, int hi) {
  const v8bf* p = (const v8bf*)(Wt + (size_t)n * K + k0 + hi * 8);
  v8bf b0 = p[0];
  v8bf b1 = p[2];
  v16bf b;
#pragma unroll
  for (int e = 0; e < 8; ++e) { b[e] = b0[e]; b[8 + e] = b1[e]; }
  return b;
}

// ---- Weight prep: Wt[n*K + k] = (bf16) W[k*N + n] -------------------------
__global__ void transpose_w_bf16(const float* __restrict__ W,
                                 __bf16* __restrict__ Wt, int K, int N) {
  int idx = blockIdx.x * blockDim.x + threadIdx.x;
  if (idx >= K * N) return;
  int k = idx / N, n = idx - k * N;          // coalesced read of W
  Wt[(size_t)n * K + k] = (__bf16)W[idx];
}

// ---- Fused kernel: ft_strip = softmax(relu((X@W+b)@Wm0+bm0)@Wm1+bm1) ------
// block = 128 threads (4 waves); one 16-row strip per block.
// Phase 1: projection GEMM, wave w -> cols [w*64, w*64+64).
// Phase 2: bias + bf16 -> LDS h-strip.
// Phase 3: layer-1 GEMM from LDS, relu+bias -> LDS t-strip (fp32).
// Phase 4: 2-classes-per-thread GEMV + 8-lane-group softmax -> FT.
__global__ void proj_mlp_fused(const float* __restrict__ X,
                               const __bf16* __restrict__ Wt,
                               const float* __restrict__ bias,
                               const __bf16* __restrict__ Wm0t,
                               const float* __restrict__ bm0,
                               const float* __restrict__ Wm1,
                               const float* __restrict__ bm1,
                               float* __restrict__ FT,
                               int K) {
  __shared__ __bf16 h_lds[16 * HPAD];   // ~8.3 KB
  __shared__ float  t_lds[16 * TPAD];   // ~16.3 KB
  const int lane = threadIdx.x & 31;
  const int wave = threadIdx.x >> 5;   // 0..3
  const int hi   = lane >> 4;
  const int ln   = lane & 15;
  const int row0 = blockIdx.x * 16;
  const float* arow = X + (size_t)(row0 + ln) * K;

  v8f acc[4];
#pragma unroll
  for (int t = 0; t < 4; ++t)
#pragma unroll
    for (int r = 0; r < 8; ++r) acc[t][r] = 0.0f;

  // ---- Phase 1: projection GEMM ----
  for (int k0 = 0; k0 < K; k0 += 32) {
    __builtin_prefetch(arow + k0 + 32, 0, 1);   // global_prefetch_b8, speculative
    v16bf a = load_a_frag_f32(arow, k0, hi);
#pragma unroll
    for (int t = 0; t < 4; ++t) {
      int n = wave * 64 + t * 16 + ln;
      v16bf b = load_b_frag_bf16(Wt, K, k0, n, hi);
      acc[t] = __builtin_amdgcn_wmma_f32_16x16x32_bf16(
          false, a, false, b, (short)0, acc[t], false, false);
    }
  }

  // ---- Phase 2: bias, bf16, stage h-strip in LDS ----
#pragma unroll
  for (int t = 0; t < 4; ++t) {
    int n = wave * 64 + t * 16 + ln;
    float bv = bias[n];
#pragma unroll
    for (int r = 0; r < 8; ++r) {
      int m = r + 8 * hi;
      h_lds[m * HPAD + n] = (__bf16)(acc[t][r] + bv);
    }
  }
  __syncthreads();

  // ---- Phase 3: layer-1 GEMM from LDS (K = HIDDEN = 256) ----
#pragma unroll
  for (int t = 0; t < 4; ++t)
#pragma unroll
    for (int r = 0; r < 8; ++r) acc[t][r] = 0.0f;

  const __bf16* hrow = h_lds + ln * HPAD;
#pragma unroll
  for (int k0 = 0; k0 < HIDDEN; k0 += 32) {
    v16bf a = load_a_frag_lds(hrow, k0, hi);
#pragma unroll
    for (int t = 0; t < 4; ++t) {
      int n = wave * 64 + t * 16 + ln;
      v16bf b = load_b_frag_bf16(Wm0t, HIDDEN, k0, n, hi);
      acc[t] = __builtin_amdgcn_wmma_f32_16x16x32_bf16(
          false, a, false, b, (short)0, acc[t], false, false);
    }
  }

#pragma unroll
  for (int t = 0; t < 4; ++t) {
    int n = wave * 64 + t * 16 + ln;
    float bv = bm0[n];
#pragma unroll
    for (int r = 0; r < 8; ++r) {
      int m = r + 8 * hi;
      t_lds[m * TPAD + n] = fmaxf(acc[t][r] + bv, 0.0f);  // relu
    }
  }
  __syncthreads();

  // ---- Phase 4: 128 threads = 16 rows x 8 class-pairs; GEMV + softmax ----
  const int row = threadIdx.x >> 3;       // 0..15
  const int c2  = threadIdx.x & 7;        // class pair index
  const int cls0 = 2 * c2, cls1 = 2 * c2 + 1;
  float lg0 = bm1[cls0];
  float lg1 = bm1[cls1];
  const float* trow = t_lds + row * TPAD;
  for (int k = 0; k < HIDDEN; ++k) {
    float tv = trow[k];
    lg0 = __builtin_fmaf(tv, Wm1[k * NUM_CLASSES + cls0], lg0);
    lg1 = __builtin_fmaf(tv, Wm1[k * NUM_CLASSES + cls1], lg1);
  }
  // softmax over 16 classes held by an 8-lane group (xor masks < 8 stay in-group)
  float mx = fmaxf(lg0, lg1);
#pragma unroll
  for (int off = 4; off; off >>= 1) mx = fmaxf(mx, __shfl_xor(mx, off, 32));
  float ex0 = __expf(lg0 - mx);
  float ex1 = __expf(lg1 - mx);
  float sm = ex0 + ex1;
#pragma unroll
  for (int off = 4; off; off >>= 1) sm += __shfl_xor(sm, off, 32);
  float inv = 1.0f / sm;
  float* fo = FT + (size_t)(row0 + row) * NUM_CLASSES;
  fo[cls0] = ex0 * inv;
  fo[cls1] = ex1 * inv;
}

// ---- Workspace init (deterministic every call) ----------------------------
__global__ void init_ws(float* __restrict__ m, float* __restrict__ z,
                        float* __restrict__ agg) {
  int i = blockIdx.x * blockDim.x + threadIdx.x;
  if (i < N_NODES) { m[i] = -3.0e38f; z[i] = 0.0f; }
  if (i < NODE_NUM * NUM_CLASSES) agg[i] = 0.0f;
}

// ---- Edge phase -----------------------------------------------------------
static __device__ __forceinline__ void atomicMaxF(float* addr, float v) {
  if (v >= 0.0f) atomicMax((int*)addr, __float_as_int(v));
  else           atomicMin((unsigned int*)addr, __float_as_uint(v));
}

__global__ void edge_max(const int* __restrict__ dst, const float* __restrict__ e,
                         float* __restrict__ m) {
  int i = blockIdx.x * blockDim.x + threadIdx.x;
  if (i >= N_EDGES) return;
  int d = dst[i];
  if (d >= NODE_NUM) return;   // only dst < NODE_NUM affects output
  atomicMaxF(&m[d], e[i]);
}

__global__ void edge_z(const int* __restrict__ dst, const float* __restrict__ e,
                       const float* __restrict__ m, float* __restrict__ z) {
  int i = blockIdx.x * blockDim.x + threadIdx.x;
  if (i >= N_EDGES) return;
  int d = dst[i];
  if (d >= NODE_NUM) return;
  atomicAdd(&z[d], __expf(e[i] - m[d]));
}

__global__ void edge_scatter(const int* __restrict__ src, const int* __restrict__ dst,
                             const float* __restrict__ e, const float* __restrict__ m,
                             const float* __restrict__ z, const float* __restrict__ ft,
                             float* __restrict__ agg) {
  int i = blockIdx.x * blockDim.x + threadIdx.x;
  if (i >= N_EDGES) return;
  int d = dst[i];
  if (d >= NODE_NUM) return;
  float w = __expf(e[i] - m[d]) / z[d];
  int s = src[i];
  const float4* fs = (const float4*)(ft + (size_t)s * NUM_CLASSES);
  float* ag = agg + (size_t)d * NUM_CLASSES;
#pragma unroll
  for (int q = 0; q < 4; ++q) {
    float4 v = fs[q];
    atomicAdd(&ag[q * 4 + 0], v.x * w);
    atomicAdd(&ag[q * 4 + 1], v.y * w);
    atomicAdd(&ag[q * 4 + 2], v.z * w);
    atomicAdd(&ag[q * 4 + 3], v.w * w);
  }
}

// ---- Final gated combine: out = [logits(40000x16); alpha(40000)] ----------
__global__ void combine(const float* __restrict__ ft, const float* __restrict__ agg,
                        const float* __restrict__ alpha, float* __restrict__ out) {
  int i = blockIdx.x * blockDim.x + threadIdx.x;
  if (i < NODE_NUM * NUM_CLASSES) {
    int node = i >> 4;
    float g = 1.0f / (1.0f + __expf(-alpha[node]));
    out[i] = g * ft[i] + (1.0f - g) * agg[i];   // self_cls == ft[:NODE_NUM]
  }
  if (i < NODE_NUM) out[NODE_NUM * NUM_CLASSES + i] = alpha[i];
}

// ---------------------------------------------------------------------------
extern "C" void kernel_launch(void* const* d_in, const int* in_sizes, int n_in,
                              void* d_out, int out_size, void* d_ws, size_t ws_size,
                              hipStream_t stream) {
  (void)in_sizes; (void)n_in; (void)out_size; (void)ws_size;
  const float* feat0 = (const float*)d_in[0];
  const float* feat1 = (const float*)d_in[1];
  const float* feat2 = (const float*)d_in[2];
  const int*   src   = (const int*)  d_in[3];
  const int*   dst   = (const int*)  d_in[4];
  const float* W0    = (const float*)d_in[5];
  const float* b0    = (const float*)d_in[6];
  const float* W1    = (const float*)d_in[7];
  const float* b1    = (const float*)d_in[8];
  const float* W2    = (const float*)d_in[9];
  const float* b2    = (const float*)d_in[10];
  const float* Wm0   = (const float*)d_in[11];
  const float* bm0   = (const float*)d_in[12];
  const float* Wm1   = (const float*)d_in[13];
  const float* bm1   = (const float*)d_in[14];
  const float* alpha = (const float*)d_in[15];
  const float* e     = (const float*)d_in[16];

  // ---- workspace layout (fp32 region, then 16B-aligned bf16 region) ------
  float* ft   = (float*)d_ws;                                  // 100000*16
  float* mbuf = ft   + (size_t)N_NODES * NUM_CLASSES;          // 100000
  float* zbuf = mbuf + N_NODES;                                // 100000
  float* agg  = zbuf + N_NODES;                                // 40000*16
  __bf16* w0t  = (__bf16*)(agg + (size_t)NODE_NUM * NUM_CLASSES); // 512*256
  __bf16* w1t  = w0t + 512 * 256;                                 // 256*256
  __bf16* w2t  = w1t + 256 * 256;                                 // 128*256
  __bf16* wm0t = w2t + 128 * 256;                                 // 256*256
  float* out  = (float*)d_out;

  // ---- weight prep: transpose + bf16 (tiny, L2-resident) -----------------
  transpose_w_bf16<<<(512 * 256 + 255) / 256, 256, 0, stream>>>(W0,  w0t,  512, HIDDEN);
  transpose_w_bf16<<<(256 * 256 + 255) / 256, 256, 0, stream>>>(W1,  w1t,  256, HIDDEN);
  transpose_w_bf16<<<(128 * 256 + 255) / 256, 256, 0, stream>>>(W2,  w2t,  128, HIDDEN);
  transpose_w_bf16<<<(256 * 256 + 255) / 256, 256, 0, stream>>>(Wm0, wm0t, 256, HIDDEN);

  // ---- fused projection + MLP + softmax (h never touches HBM) ------------
  proj_mlp_fused<<<40000 / 16, 128, 0, stream>>>(feat0, w0t, b0, wm0t, bm0,
                                                 Wm1, bm1, ft, 512);
  proj_mlp_fused<<<30000 / 16, 128, 0, stream>>>(feat1, w1t, b1, wm0t, bm0,
                                                 Wm1, bm1,
                                                 ft + (size_t)40000 * NUM_CLASSES, 256);
  proj_mlp_fused<<<30000 / 16, 128, 0, stream>>>(feat2, w2t, b2, wm0t, bm0,
                                                 Wm1, bm1,
                                                 ft + (size_t)70000 * NUM_CLASSES, 128);

  // ---- edge softmax + scatter (only dst < NODE_NUM contributes) ----------
  init_ws<<<(NODE_NUM * NUM_CLASSES + 255) / 256, 256, 0, stream>>>(mbuf, zbuf, agg);
  edge_max    <<<(N_EDGES + 255) / 256, 256, 0, stream>>>(dst, e, mbuf);
  edge_z      <<<(N_EDGES + 255) / 256, 256, 0, stream>>>(dst, e, mbuf, zbuf);
  edge_scatter<<<(N_EDGES + 255) / 256, 256, 0, stream>>>(src, dst, e, mbuf, zbuf,
                                                          ft, agg);

  // ---- gated combine + alpha passthrough ---------------------------------
  combine<<<(NODE_NUM * NUM_CLASSES + 255) / 256, 256, 0, stream>>>(ft, agg, alpha, out);
}
